// DualAttentionLayerO2TwoUpdateNodeGeneral_1692217115461
// MI455X (gfx1250) — matrix operand order repaired and checked by reading, not compiled, not run on hardware
//
#include <hip/hip_runtime.h>
#include <hip/hip_bf16.h>

// -------------------------------------------------------------------------
// DualAttentionLayer (BindDM x2h/h2x) for MI455X (gfx1250, wave32, WMMA).
// Heavy edge MLPs run as f16 WMMA (v_wmma_f32_16x16x32_f16) with f32
// accumulate. W1 weight fragments are staged in LDS once per workgroup and
// shared by all 8 waves (B-fragments then come from ds_load_b128 instead of
// re-streaming ~120KB of weights from L2 per 16-row tile). Waves grid-stride
// over row tiles to amortize the staging. Scatter softmax via float atomics.
// -------------------------------------------------------------------------

typedef __attribute__((ext_vector_type(16))) _Float16 v16h;
typedef __attribute__((ext_vector_type(8)))  _Float16 v8h;
typedef __attribute__((ext_vector_type(8)))  float    v8f;

#define HD_   128   // hidden dim
#define NH_   16    // heads
#define DH_   8     // dim per head
#define EF_   4     // edge_attr features
#define NG_   20    // gaussians
#define KV_   340   // kv feature dim = 4 + 80 + 128 + 128
#define KVP_  352   // padded to multiple of 32

// ---------------- fragment loaders (CDNA5 WMMA 16x16x32 f16 layouts) -----

__device__ __forceinline__ v16h make_v16(v8h lo, v8h hi) {
  return __builtin_shufflevector(lo, hi, 0, 1, 2, 3, 4, 5, 6, 7,
                                 8, 9, 10, 11, 12, 13, 14, 15);
}

// A 16x32 f16: lane row = lane&15; halves 0..7 -> K = k0 + (lane<16?0:8)+h,
// halves 8..15 -> K = k0 + 16 + (lane<16?0:8) + (h-8). Two aligned 16B loads.
__device__ __forceinline__ v16h load_a16(const _Float16* __restrict__ rowp,
                                         int k0, int lane) {
  const _Float16* p = rowp + k0 + ((lane & 16) ? 8 : 0);
  v8h lo = *(const v8h*)p;
  v8h hi = *(const v8h*)(p + 16);
  return make_v16(lo, hi);
}

// B 32x16 f16 pre-packed per fragment: 512 halfs, lane-major (lane*16 + h).
// Works for both global and LDS source pointers (inlined -> ds_load_b128
// when the pointer derives from __shared__).
__device__ __forceinline__ v16h load_bfrag(const _Float16* w, int frag,
                                           int lane) {
  const _Float16* p = w + ((size_t)frag << 9) + (lane << 4);
  const v8h* pv = (const v8h*)p;
  return make_v16(pv[0], pv[1]);
}

// ---------------- weight packing:  W[k][n] f32 -> B-fragment f16 ---------
// frag (kt,nt): col=nt*16+(lane&15); h<8 -> K=kt*32+(lane<16?0:8)+h, h>=8 -> K+=16-8.
__global__ void pack_b_frags(const float* __restrict__ W,
                             _Float16* __restrict__ outp,
                             int Kp, int Kr, int Nc, int total) {
  int idx = blockIdx.x * blockDim.x + threadIdx.x;
  if (idx >= total) return;
  int frag = idx >> 9;
  int w    = idx & 511;
  int lane = w >> 4;
  int h    = w & 15;
  int ntper = Nc / 16;
  int kt = frag / ntper, nt = frag - kt * ntper;
  int k = kt * 32 + ((lane & 16) ? 8 : 0) + (h & 7) + ((h & 8) ? 16 : 0);
  int n = nt * 16 + (lane & 15);
  float v = (k < Kr) ? W[(size_t)k * Nc + n] : 0.0f;
  outp[idx] = (_Float16)v;
}

// ---------------- fused 2-layer MLP (Linear/LN/ReLU/Linear) via WMMA -----
template <int KPAD, int ODIM, bool TWO_LAYER, bool OUT_F16>
__global__ __launch_bounds__(256) void mlp_wmma_kernel(
    const _Float16* __restrict__ in16,
    const _Float16* __restrict__ w1f, const float* __restrict__ b1,
    const float* __restrict__ gam, const float* __restrict__ bet,
    const _Float16* __restrict__ w2f, const float* __restrict__ b2,
    const float* __restrict__ rowscale, const float* __restrict__ residual,
    float* __restrict__ out32, _Float16* __restrict__ out16, int M) {
  constexpr int NT1 = (TWO_LAYER ? HD_ : ODIM) / 16;
  constexpr int KT1 = KPAD / 32;
  constexpr int NTO = ODIM / 16;
  constexpr int W1H = KT1 * NT1 * 512;  // halfs of staged W1 fragments

  extern __shared__ __align__(16) _Float16 smem[];
  _Float16* w1s = smem;            // [W1H]
  _Float16* hid = smem + W1H;      // 8 waves * 16*128 halfs (TWO_LAYER only)

  // cooperative staging of W1 fragments into LDS (shared by all 8 waves)
  {
    const v8h* s8 = (const v8h*)w1f;
    v8h* d8 = (v8h*)w1s;
    for (int i = threadIdx.x; i < W1H / 8; i += 256) d8[i] = s8[i];
  }
  __syncthreads();

  const int lane = threadIdx.x & 31;
  const int wave = threadIdx.x >> 5;
  const int rb = (lane & 16) ? 8 : 0;
  const int tstride = gridDim.x * 8;

  for (int tile = blockIdx.x * 8 + wave; tile * 16 < M; tile += tstride) {
    const int m0 = tile * 16;
    int rA = m0 + (lane & 15);
    if (rA > M - 1) rA = M - 1;
    const _Float16* arow = in16 + (size_t)rA * KPAD;

    v8f acc[NT1];
#pragma unroll
    for (int t = 0; t < NT1; ++t)
#pragma unroll
      for (int i = 0; i < 8; ++i) acc[t][i] = 0.0f;

#pragma unroll
    for (int kt = 0; kt < KT1; ++kt) {
      v16h a = load_a16(arow, kt * 32, lane);
#pragma unroll
      for (int t = 0; t < NT1; ++t) {
        v16h b = load_bfrag(w1s, kt * NT1 + t, lane);  // from LDS
        acc[t] = __builtin_amdgcn_wmma_f32_16x16x32_f16(false, a, false, b,
                                                        (short)0, acc[t],
                                                        false, false);
      }
    }

    v8f fin[NTO];
    const float* bias;

    if constexpr (TWO_LAYER) {
      // LayerNorm over 128 hidden features of each of this tile's 16 rows.
      // C-layout: lanes 0-15 hold rows 0-7 (reg r), lanes 16-31 rows 8-15.
      float mean[8], rstd[8];
#pragma unroll
      for (int r = 0; r < 8; ++r) {
        float s1 = 0.f, s2 = 0.f;
#pragma unroll
        for (int t = 0; t < NT1; ++t) {
          float v = acc[t][r];
          s1 += v;
          s2 += v * v;
        }
        for (int msk = 8; msk >= 1; msk >>= 1) {  // reduce over 16-lane half
          s1 += __shfl_xor(s1, msk, 32);
          s2 += __shfl_xor(s2, msk, 32);
        }
        float mu  = s1 * (1.0f / 128.0f);
        float var = s2 * (1.0f / 128.0f) - mu * mu;
        mean[r] = mu;
        rstd[r] = __frsqrt_rn(var + 1e-5f);
      }
      _Float16* lh = hid + wave * (16 * HD_);
#pragma unroll
      for (int t = 0; t < NT1; ++t) {
        int col = t * 16 + (lane & 15);
        float gc = gam[col], bc = bet[col];
#pragma unroll
        for (int r = 0; r < 8; ++r) {
          float v = (acc[t][r] - mean[r]) * rstd[r] * gc + bc;
          lh[(rb + r) * HD_ + col] = (_Float16)(v > 0.f ? v : 0.f);
        }
      }
      asm volatile("s_wait_dscnt 0" ::: "memory");

      v8f acc2[NTO];
#pragma unroll
      for (int t = 0; t < NTO; ++t)
#pragma unroll
        for (int i = 0; i < 8; ++i) acc2[t][i] = 0.0f;
      const _Float16* arow2 = lh + (lane & 15) * HD_;
#pragma unroll
      for (int kt = 0; kt < HD_ / 32; ++kt) {
        v16h a2 = load_a16(arow2, kt * 32, lane);
#pragma unroll
        for (int t = 0; t < NTO; ++t) {
          v16h b = load_bfrag(w2f, kt * NTO + t, lane);  // small, from L2
          acc2[t] = __builtin_amdgcn_wmma_f32_16x16x32_f16(
              false, a2, false, b, (short)0, acc2[t], false, false);
        }
      }
#pragma unroll
      for (int t = 0; t < NTO; ++t) fin[t] = acc2[t];
      bias = b2;
    } else {
#pragma unroll
      for (int t = 0; t < NTO; ++t) fin[t] = acc[t];
      bias = b1;
    }

#pragma unroll
    for (int t = 0; t < NTO; ++t) {
      int col = t * 16 + (lane & 15);
      float bb = bias[col];
#pragma unroll
      for (int r = 0; r < 8; ++r) {
        int row = m0 + rb + r;
        if (row < M) {
          float v = fin[t][r] + bb;
          if (rowscale) v *= rowscale[row];
          if (residual) v += residual[(size_t)row * ODIM + col];
          if constexpr (OUT_F16)
            out16[(size_t)row * ODIM + col] = (_Float16)v;
          else
            out32[(size_t)row * ODIM + col] = v;
        }
      }
    }
  }
}

// ---------------- per-edge feature assembly ------------------------------
__global__ void edge_prep_kernel(
    const float* __restrict__ xpos, const float* __restrict__ hfeat,
    const float* __restrict__ ea, const int* __restrict__ srcI,
    const int* __restrict__ dstI, const float* __restrict__ ewW,
    const float* __restrict__ ewb, float* __restrict__ ew_out,
    _Float16* __restrict__ kv16, float* __restrict__ relx, int E) {
  int e = blockIdx.x * blockDim.x + threadIdx.x;
  if (e >= E) return;
  int s = srcI[e], d = dstI[e];
  float rx = xpos[(size_t)d * 3 + 0] - xpos[(size_t)s * 3 + 0];
  float ry = xpos[(size_t)d * 3 + 1] - xpos[(size_t)s * 3 + 1];
  float rz = xpos[(size_t)d * 3 + 2] - xpos[(size_t)s * 3 + 2];
  float dist = sqrtf(rx * rx + ry * ry + rz * rz);
  if (relx) {
    relx[(size_t)e * 3 + 0] = rx;
    relx[(size_t)e * 3 + 1] = ry;
    relx[(size_t)e * 3 + 2] = rz;
  }
  const float step = 10.0f / 19.0f;
  const float coeff = -0.5f / (step * step);
  float gs[NG_];
#pragma unroll
  for (int j = 0; j < NG_; ++j) {
    float t = dist - step * j;
    gs[j] = expf(coeff * t * t);
  }
  _Float16* kv = kv16 + (size_t)e * KVP_;
  float eav[EF_];
#pragma unroll
  for (int i = 0; i < EF_; ++i) {
    eav[i] = ea[(size_t)e * EF_ + i];
    kv[i] = (_Float16)eav[i];
  }
  float da = 0.f;
#pragma unroll
  for (int i = 0; i < EF_; ++i)
#pragma unroll
    for (int j = 0; j < NG_; ++j) {
      int c = i * NG_ + j;
      float rf = eav[i] * gs[j];
      kv[EF_ + c] = (_Float16)rf;
      da += rf * ewW[c];
    }
  ew_out[e] = 1.0f / (1.0f + expf(-(da + ewb[0])));
  const float* hd = hfeat + (size_t)d * HD_;
  const float* hs = hfeat + (size_t)s * HD_;
#pragma unroll 8
  for (int c = 0; c < HD_; ++c) kv[84 + c] = (_Float16)hd[c];
#pragma unroll 8
  for (int c = 0; c < HD_; ++c) kv[84 + HD_ + c] = (_Float16)hs[c];
#pragma unroll
  for (int c = KV_; c < KVP_; ++c) kv[c] = (_Float16)0.0f;
}

// ---------------- scatter softmax / aggregation --------------------------
__device__ __forceinline__ void atomicMaxFloat(float* addr, float val) {
  if (val >= 0.0f)
    atomicMax((int*)addr, __float_as_int(val));
  else
    atomicMin((unsigned int*)addr, __float_as_uint(val));
}

__global__ void logits_max_kernel(const _Float16* __restrict__ k16,
                                  const float* __restrict__ q,
                                  const int* __restrict__ dstI,
                                  float* __restrict__ logits,
                                  float* __restrict__ mx, int E) {
  int idx = blockIdx.x * blockDim.x + threadIdx.x;
  if (idx >= E * NH_) return;
  int e = idx >> 4, hh = idx & 15;
  int d = dstI[e];
  const _Float16* kp = k16 + (size_t)e * HD_ + hh * DH_;
  const float* qp = q + (size_t)d * HD_ + hh * DH_;
  float sdot = 0.f;
#pragma unroll
  for (int i = 0; i < DH_; ++i) sdot += qp[i] * (float)kp[i];
  sdot *= 0.35355339059327373f;  // 1/sqrt(8)
  logits[idx] = sdot;
  atomicMaxFloat(mx + (size_t)d * NH_ + hh, sdot);
}

__global__ void expsum_kernel(const int* __restrict__ dstI,
                              const float* __restrict__ mx,
                              float* __restrict__ lg,
                              float* __restrict__ ssum, int E) {
  int idx = blockIdx.x * blockDim.x + threadIdx.x;
  if (idx >= E * NH_) return;
  int e = idx >> 4, hh = idx & 15;
  int d = dstI[e];
  float ex = expf(lg[idx] - mx[(size_t)d * NH_ + hh]);
  lg[idx] = ex;
  atomicAdd(ssum + (size_t)d * NH_ + hh, ex);
}

__global__ void accum_v_kernel(const int* __restrict__ dstI,
                               const float* __restrict__ ex,
                               const float* __restrict__ ssum,
                               const _Float16* __restrict__ v16,
                               float* __restrict__ attn, long total) {
  long idx = blockIdx.x * (long)blockDim.x + threadIdx.x;
  if (idx >= total) return;
  int e = (int)(idx >> 7), c = (int)(idx & 127);
  int hh = c >> 3;
  int d = dstI[e];
  float alpha = ex[(size_t)e * NH_ + hh] / (ssum[(size_t)d * NH_ + hh] + 1e-16f);
  atomicAdd(attn + (size_t)d * HD_ + c, alpha * (float)v16[idx]);
}

__global__ void accum_dx_kernel(const int* __restrict__ dstI,
                                const float* __restrict__ ex,
                                const float* __restrict__ ssum,
                                const float* __restrict__ xv,
                                const float* __restrict__ relx,
                                float* __restrict__ delta, int E) {
  int idx = blockIdx.x * blockDim.x + threadIdx.x;
  if (idx >= E * NH_) return;
  int e = idx >> 4, hh = idx & 15;
  int d = dstI[e];
  float alpha = ex[idx] / (ssum[(size_t)d * NH_ + hh] + 1e-16f);
  float w = alpha * xv[idx] * (1.0f / 16.0f);  // mean over heads
  atomicAdd(delta + (size_t)d * 3 + 0, w * relx[(size_t)e * 3 + 0]);
  atomicAdd(delta + (size_t)d * 3 + 1, w * relx[(size_t)e * 3 + 1]);
  atomicAdd(delta + (size_t)d * 3 + 2, w * relx[(size_t)e * 3 + 2]);
}

// ---------------- small utility kernels ----------------------------------
__global__ void fill_kernel(float* __restrict__ p, float v, long n) {
  long i = blockIdx.x * (long)blockDim.x + threadIdx.x;
  if (i < n) p[i] = v;
}

__global__ void cast_kernel(const float* __restrict__ s,
                            _Float16* __restrict__ dp, long n) {
  long i = blockIdx.x * (long)blockDim.x + threadIdx.x;
  if (i < n) dp[i] = (_Float16)s[i];
}

__global__ void concat2_kernel(const float* __restrict__ a,
                               const float* __restrict__ b,
                               _Float16* __restrict__ o, int M) {
  int idx = blockIdx.x * blockDim.x + threadIdx.x;
  if (idx >= M * 256) return;
  int row = idx >> 8, c = idx & 255;
  float v = (c < 128) ? a[(size_t)row * 128 + c]
                      : (b ? b[(size_t)row * 128 + (c - 128)] : 0.0f);
  o[idx] = (_Float16)v;
}

__global__ void scatter_sub_kernel(const int* __restrict__ perm,
                                   const float* __restrict__ hs,
                                   _Float16* __restrict__ o, long total) {
  long idx = blockIdx.x * (long)blockDim.x + threadIdx.x;
  if (idx >= total) return;
  int i = (int)(idx >> 7), c = (int)(idx & 127);
  o[(size_t)perm[i] * 256 + 128 + c] = (_Float16)hs[idx];
}

__global__ void xnew_kernel(const float* __restrict__ x,
                            const float* __restrict__ delta,
                            const float* __restrict__ mask,
                            float* __restrict__ o, int Nn) {
  int idx = blockIdx.x * blockDim.x + threadIdx.x;
  if (idx >= Nn * 3) return;
  int r = idx / 3;
  o[idx] = x[idx] + delta[idx] * mask[r];
}

// -------------------------------------------------------------------------
// Host orchestration.
// Input order (setup_inputs insertion order; params flattened recursively in
// insertion order):
//  0 h  1 h_sub  2 x  3 x_sub  4 edge_attr  5 edge_attr_sub
//  6 edge_index  7 edge_index_sub  8 mask_ligand  9 perm_sub
//  10.. params: x2h{hk(W1,b1,g,beta,W2,b2) hv hq ew_W ew_b no} = 10..35
//               x2h_sub same = 36..61
//               h2x{xk xv xq ew_W ew_b} = 62..81, fus_W=82, fus_b=83
// -------------------------------------------------------------------------
extern "C" void kernel_launch(void* const* d_in, const int* in_sizes, int n_in,
                              void* d_out, int out_size, void* d_ws,
                              size_t ws_size, hipStream_t stream) {
  if (n_in < 84) return;
  auto F = [&](int i) { return (const float*)d_in[i]; };
  auto Ii = [&](int i) { return (const int*)d_in[i]; };

  const int N  = in_sizes[0] / HD_;
  const int NS = in_sizes[1] / HD_;
  const int E  = in_sizes[4] / EF_;
  const int ES = in_sizes[5] / EF_;

  const float* h      = F(0);
  const float* h_sub  = F(1);
  const float* x      = F(2);
  const float* x_sub  = F(3);
  const float* ea     = F(4);
  const float* ea_sub = F(5);
  const int* src   = Ii(6);
  const int* dst   = Ii(6) + E;
  const int* src_s = Ii(7);
  const int* dst_s = Ii(7) + ES;
  const float* mask = F(8);
  const int* perm   = Ii(9);

  size_t off = 0;
  auto alloc = [&](size_t bytes) {
    off = (off + 255) & ~(size_t)255;
    void* p = (char*)d_ws + off;
    off += bytes;
    return p;
  };
  auto packW = [&](const float* W, int Kp, int Kr, int Nc) {
    size_t halfs = (size_t)(Kp / 32) * (Nc / 16) * 512;
    _Float16* dp = (_Float16*)alloc(halfs * sizeof(_Float16));
    int total = (int)halfs;
    pack_b_frags<<<(total + 255) / 256, 256, 0, stream>>>(W, dp, Kp, Kr, Nc,
                                                          total);
    return dp;
  };

  // weights -> f16 fragment layout (repacked every call; ~1 MB total)
  _Float16* hkW1  = packW(F(10), KVP_, KV_, HD_);
  _Float16* hkW2  = packW(F(14), HD_, HD_, HD_);
  _Float16* hvW1  = packW(F(16), KVP_, KV_, HD_);
  _Float16* hvW2  = packW(F(20), HD_, HD_, HD_);
  _Float16* hqW1  = packW(F(22), HD_, HD_, HD_);
  _Float16* hqW2  = packW(F(26), HD_, HD_, HD_);
  _Float16* noW1  = packW(F(30), 256, 256, HD_);
  _Float16* noW2  = packW(F(34), HD_, HD_, HD_);
  _Float16* hkW1s = packW(F(36), KVP_, KV_, HD_);
  _Float16* hkW2s = packW(F(40), HD_, HD_, HD_);
  _Float16* hvW1s = packW(F(42), KVP_, KV_, HD_);
  _Float16* hvW2s = packW(F(46), HD_, HD_, HD_);
  _Float16* hqW1s = packW(F(48), HD_, HD_, HD_);
  _Float16* hqW2s = packW(F(52), HD_, HD_, HD_);
  _Float16* noW1s = packW(F(56), 256, 256, HD_);
  _Float16* noW2s = packW(F(60), HD_, HD_, HD_);
  _Float16* xkW1  = packW(F(62), KVP_, KV_, HD_);
  _Float16* xkW2  = packW(F(66), HD_, HD_, HD_);
  _Float16* xvW1  = packW(F(68), KVP_, KV_, HD_);
  _Float16* xvW2  = packW(F(72), HD_, HD_, NH_);
  _Float16* xqW1  = packW(F(74), HD_, HD_, HD_);
  _Float16* xqW2  = packW(F(78), HD_, HD_, HD_);
  _Float16* fusW  = packW(F(82), 256, 256, HD_);

  // scratch buffers (sub stages reuse the main-sized ones)
  _Float16* kv16  = (_Float16*)alloc((size_t)E * KVP_ * 2);
  _Float16* k16   = (_Float16*)alloc((size_t)E * HD_ * 2);
  _Float16* v16   = (_Float16*)alloc((size_t)E * HD_ * 2);
  float* exbuf    = (float*)alloc((size_t)E * NH_ * 4);
  float* xvbuf    = (float*)alloc((size_t)E * NH_ * 4);
  float* ew_a     = (float*)alloc((size_t)E * 4);
  float* relx     = (float*)alloc((size_t)E * 3 * 4);
  float* qbuf     = (float*)alloc((size_t)N * HD_ * 4);
  float* mbuf     = (float*)alloc((size_t)N * NH_ * 4);
  float* sbuf     = (float*)alloc((size_t)N * NH_ * 4);
  float* attn     = (float*)alloc((size_t)N * HD_ * 4);
  float* h_in     = (float*)alloc((size_t)N * HD_ * 4);
  float* h_in_s   = (float*)alloc((size_t)NS * HD_ * 4);
  float* delta    = (float*)alloc((size_t)N * 3 * 4);
  _Float16* node16 = (_Float16*)alloc((size_t)N * 256 * 2);
  _Float16* h16b   = (_Float16*)alloc((size_t)N * HD_ * 2);

  auto fillf = [&](float* p, float v, long n) {
    fill_kernel<<<(int)((n + 255) / 256), 256, 0, stream>>>(p, v, n);
  };
  auto castf = [&](const float* s, _Float16* dp, long n) {
    cast_kernel<<<(int)((n + 255) / 256), 256, 0, stream>>>(s, dp, n);
  };
  const int GRID_CAP = 512;  // grid-stride tile loop; amortizes LDS staging
  auto mlp_grid = [&](int M) {
    int b = ((M + 15) / 16 + 7) / 8;
    return b < GRID_CAP ? b : GRID_CAP;
  };
  // dynamic LDS: staged W1 fragments + per-wave hidden staging
  auto shmem2 = [](int KT1) { return (size_t)(KT1 * 8 * 512 + 8 * 2048) * 2; };
  const size_t SH_352 = shmem2(KVP_ / 32);   // 122880 B
  const size_t SH_256 = shmem2(256 / 32);    //  98304 B
  const size_t SH_128 = shmem2(HD_ / 32);    //  65536 B
  const size_t SH_FUS = (size_t)(8 * 8 * 512) * 2;  // 65536 B, no hidden

  // ---------------- one x2h attention block ------------------------------
  auto x2h_stage = [&](const float* hN, const float* xN, const float* eaE,
                       const int* sE, const int* dE, int n, int e,
                       _Float16* kW1, _Float16* kW2, _Float16* vW1,
                       _Float16* vW2, _Float16* qW1, _Float16* qW2,
                       _Float16* nW1, _Float16* nW2, int pb,
                       const float* ewW, const float* ewb, float* relxOut,
                       float* hinOut) {
    edge_prep_kernel<<<(e + 255) / 256, 256, 0, stream>>>(
        xN, hN, eaE, sE, dE, ewW, ewb, ew_a, kv16, relxOut, e);
    castf(hN, h16b, (long)n * HD_);
    mlp_wmma_kernel<HD_, HD_, true, false>
        <<<mlp_grid(n), 256, SH_128, stream>>>(
            h16b, qW1, F(pb + 13), F(pb + 14), F(pb + 15), qW2, F(pb + 17),
            nullptr, nullptr, qbuf, nullptr, n);
    mlp_wmma_kernel<KVP_, HD_, true, true>
        <<<mlp_grid(e), 256, SH_352, stream>>>(
            kv16, kW1, F(pb + 1), F(pb + 2), F(pb + 3), kW2, F(pb + 5),
            nullptr, nullptr, nullptr, k16, e);
    mlp_wmma_kernel<KVP_, HD_, true, true>
        <<<mlp_grid(e), 256, SH_352, stream>>>(
            kv16, vW1, F(pb + 7), F(pb + 8), F(pb + 9), vW2, F(pb + 11), ew_a,
            nullptr, nullptr, v16, e);
    fillf(mbuf, -3.0e38f, (long)n * NH_);
    fillf(sbuf, 0.0f, (long)n * NH_);
    logits_max_kernel<<<(e * NH_ + 255) / 256, 256, 0, stream>>>(
        k16, qbuf, dE, exbuf, mbuf, e);
    expsum_kernel<<<(e * NH_ + 255) / 256, 256, 0, stream>>>(dE, mbuf, exbuf,
                                                             sbuf, e);
    fillf(attn, 0.0f, (long)n * HD_);
    accum_v_kernel<<<(int)(((long)e * HD_ + 255) / 256), 256, 0, stream>>>(
        dE, exbuf, sbuf, v16, attn, (long)e * HD_);
    concat2_kernel<<<(n * 256 + 255) / 256, 256, 0, stream>>>(attn, hN,
                                                              node16, n);
    mlp_wmma_kernel<256, HD_, true, false>
        <<<mlp_grid(n), 256, SH_256, stream>>>(
            node16, nW1, F(pb + 21), F(pb + 22), F(pb + 23), nW2, F(pb + 25),
            nullptr, hN, hinOut, nullptr, n);
  };

  // main-graph x2h
  x2h_stage(h, x, ea, src, dst, N, E, hkW1, hkW2, hvW1, hvW2, hqW1, hqW2,
            noW1, noW2, 10, F(28), F(29), relx, h_in);
  // sub-graph x2h
  x2h_stage(h_sub, x_sub, ea_sub, src_s, dst_s, NS, ES, hkW1s, hkW2s, hvW1s,
            hvW2s, hqW1s, hqW2s, noW1s, noW2s, 36, F(54), F(55), nullptr,
            h_in_s);

  // fuse: x2h_out = concat([h_in, pad(h_in_sub)]) @ fus_W + fus_b
  concat2_kernel<<<(N * 256 + 255) / 256, 256, 0, stream>>>(h_in, nullptr,
                                                            node16, N);
  scatter_sub_kernel<<<(int)(((long)NS * HD_ + 255) / 256), 256, 0, stream>>>(
      perm, h_in_s, node16, (long)NS * HD_);
  float* x2h_out = (float*)d_out;  // output part 1: [N,128]
  mlp_wmma_kernel<256, HD_, false, false>
      <<<mlp_grid(N), 256, SH_FUS, stream>>>(
          node16, fusW, F(83), nullptr, nullptr, nullptr, nullptr, nullptr,
          nullptr, x2h_out, nullptr, N);

  // h2x on main graph with h = x2h_out
  castf(x2h_out, h16b, (long)N * HD_);
  mlp_wmma_kernel<HD_, HD_, true, false>
      <<<mlp_grid(N), 256, SH_128, stream>>>(
          h16b, xqW1, F(75), F(76), F(77), xqW2, F(79), nullptr, nullptr,
          qbuf, nullptr, N);
  edge_prep_kernel<<<(E + 255) / 256, 256, 0, stream>>>(
      x, x2h_out, ea, src, dst, F(80), F(81), ew_a, kv16, nullptr, E);
  mlp_wmma_kernel<KVP_, HD_, true, true>
      <<<mlp_grid(E), 256, SH_352, stream>>>(
          kv16, xkW1, F(63), F(64), F(65), xkW2, F(67), nullptr, nullptr,
          nullptr, k16, E);
  mlp_wmma_kernel<KVP_, NH_, true, false>
      <<<mlp_grid(E), 256, SH_352, stream>>>(
          kv16, xvW1, F(69), F(70), F(71), xvW2, F(73), ew_a, nullptr, xvbuf,
          nullptr, E);
  fillf(mbuf, -3.0e38f, (long)N * NH_);
  fillf(sbuf, 0.0f, (long)N * NH_);
  logits_max_kernel<<<(E * NH_ + 255) / 256, 256, 0, stream>>>(k16, qbuf, dst,
                                                               exbuf, mbuf, E);
  expsum_kernel<<<(E * NH_ + 255) / 256, 256, 0, stream>>>(dst, mbuf, exbuf,
                                                           sbuf, E);
  fillf(delta, 0.0f, (long)N * 3);
  accum_dx_kernel<<<(E * NH_ + 255) / 256, 256, 0, stream>>>(
      dst, exbuf, sbuf, xvbuf, relx, delta, E);
  xnew_kernel<<<(N * 3 + 255) / 256, 256, 0, stream>>>(
      x, delta, mask, x2h_out + (size_t)N * HD_, N);
}
